// NVTXWrappedMHA_30451318128972
// MI455X (gfx1250) — compile-verified, matrix operand order checked by reading, not run on hardware
//
#include <hip/hip_runtime.h>
#include <hip/hip_bf16.h>
#include <math.h>
#include <stdint.h>

// Problem constants (match the reference).
#define D_MODEL 2048
#define SEQ     2048
#define BATCH   2
#define NHEADS  16
#define HDIM    128
#define BSROWS  (BATCH * SEQ)   // 4096 flattened (b, s) rows

typedef __attribute__((ext_vector_type(16))) __bf16 v16bf;
typedef __attribute__((ext_vector_type(8)))  __bf16 v8bf;
typedef __attribute__((ext_vector_type(8)))  float  v8f;
typedef __attribute__((ext_vector_type(4)))  unsigned int u32x4;
typedef __attribute__((ext_vector_type(8)))  int  i32x8;
typedef __attribute__((ext_vector_type(4)))  int  i32x4;

static __device__ __forceinline__ v8f vzero8() {
  v8f z;
#pragma unroll
  for (int i = 0; i < 8; ++i) z[i] = 0.0f;
  return z;
}

// ---------------------------------------------------------------------------
// WMMA fragment helpers.
// CDNA5 16-bit A/B layout: lanes 0-15 hold K = [0..7] and [16..23];
// lanes 16-31 hold K = [8..15] and [24..31] (per-lane two contiguous 16B runs).
// Works for both global and LDS source pointers.
// ---------------------------------------------------------------------------
static __device__ __forceinline__ v16bf load_frag_k32(const __bf16* row, int k0, int hi) {
  const v8bf lo = *(const v8bf*)(row + k0 + hi * 8);
  const v8bf hh = *(const v8bf*)(row + k0 + 16 + hi * 8);
  v16bf r;
#pragma unroll
  for (int i = 0; i < 8; ++i) { r[i] = lo[i]; r[i + 8] = hh[i]; }
  return r;
}

static __device__ __forceinline__ v8f wmma_bf16(v16bf a, v16bf b, v8f c) {
  return __builtin_amdgcn_wmma_f32_16x16x32_bf16(
      /*neg_a=*/false, a, /*neg_b=*/false, b,
      /*c_mod=*/(short)0, c, /*reuse_a=*/false, /*reuse_b=*/false);
}

// ---------------------------------------------------------------------------
// Tensor Data Mover: stage a 2D bf16 tile (tile_d1 rows x tile_d0 elems) from
// global [tensor_d1 x tensor_d0] (row-major, stride0 elems) into LDS at
// lds_byte.  D# built per ISA ch.8: data_size=1 (2B); LDS padding adds
// 4 DWORDs after every 2^(pad_code+1) DWORDs stored, so tile rows land on a
// bank-conflict-free LDS pitch for the 16-row fragment reads that follow.
// ---------------------------------------------------------------------------
static __device__ __forceinline__ void tdm_load_2d(
    unsigned lds_byte, const void* gptr,
    unsigned tensor_d0, unsigned tensor_d1,
    unsigned tile_d0, unsigned tile_d1, unsigned stride0, unsigned pad_code) {
  const unsigned long long ga = (unsigned long long)(uintptr_t)gptr;
  u32x4 g0;
  g0[0] = 1u;                                    // count=1, user descriptor
  g0[1] = lds_byte;                              // lds_addr
  g0[2] = (unsigned)ga;                          // global_addr[31:0]
  g0[3] = ((unsigned)(ga >> 32) & 0x01FFFFFFu)   // global_addr[56:32]
          | 0x80000000u;                         // type=2 ("image")
  i32x8 g1;
  g1[0] = (int)((1u << 16)                       // data_size = 1 (2 bytes)
                | (1u << 20)                     // pad_enable
                | (pad_code << 22)               // pad_interval
                | (3u << 25));                   // pad_amount: 4 DWORDs
  g1[1] = (int)((tensor_d0 & 0xFFFFu) << 16);    // tensor_dim0[15:0] @ bit48
  g1[2] = (int)((tensor_d0 >> 16) | ((tensor_d1 & 0xFFFFu) << 16));
  g1[3] = (int)((tensor_d1 >> 16) | (tile_d0 << 16));
  g1[4] = (int)(tile_d1 & 0xFFFFu);              // tile_dim1; tile_dim2=0
  g1[5] = (int)stride0;                          // tensor_dim0_stride[31:0]
  g1[6] = 0;                                     // stride0[47:32], stride1 lo
  g1[7] = 0;                                     // stride1 hi
  i32x4 z4; z4[0] = 0; z4[1] = 0; z4[2] = 0; z4[3] = 0;
  i32x8 z8;
#pragma unroll
  for (int i = 0; i < 8; ++i) z8[i] = 0;
  __builtin_amdgcn_tensor_load_to_lds(g0, g1, z4, z4, z8, 0);
}

// ---------------------------------------------------------------------------
// 1) fp32 -> bf16 conversion (bandwidth pass).
// ---------------------------------------------------------------------------
__global__ __launch_bounds__(256) void cvt_f32_bf16(const float* __restrict__ s,
                                                    __bf16* __restrict__ d, int n) {
  int i = blockIdx.x * 256 + threadIdx.x;
  if (i < n) d[i] = (__bf16)s[i];
}

// ---------------------------------------------------------------------------
// Shared TN-GEMM mainloop with TDM double-buffered LDS staging.
// Block tile: M=128 (8 waves x 16 rows) x N=64; K stepped by 32.
// ---------------------------------------------------------------------------
#define LDS_PITCH  40                        // elems per LDS tile row (64B+16B pad)
#define ABUF_ELEMS (128 * LDS_PITCH)         // 5120
#define BBUF_ELEMS (64 * LDS_PITCH)          // 2560
#define BUF_ELEMS  (ABUF_ELEMS + BBUF_ELEMS) // 7680 elems (15360 B) per buffer

static __device__ __forceinline__ void gemm_tn_mainloop(
    const __bf16* __restrict__ A, const __bf16* __restrict__ B,
    int m0blk, int n0, int arows, int brows,
    __bf16* lds, int lane, int w, v8f acc[4]) {
  const int hi = lane >> 4, ln = lane & 15;
  const unsigned lds0 = (unsigned)(uintptr_t)lds;   // LDS byte address (low 32b)

  if (w == 0) {   // prologue: stage K-step 0 into buffer 0
    tdm_load_2d(lds0, A + (size_t)m0blk * D_MODEL,
                D_MODEL, (unsigned)arows, 32, 128, D_MODEL, 3);
    tdm_load_2d(lds0 + ABUF_ELEMS * 2, B + (size_t)n0 * D_MODEL,
                D_MODEL, (unsigned)brows, 32, 64, D_MODEL, 3);
  }

  int buf = 0;
  for (int k0 = 0; k0 < D_MODEL; k0 += 32) {
    if (k0 + 32 < D_MODEL) {
      const int nb = buf ^ 1;
      if (w == 0) {   // prefetch next K-step into the other buffer
        tdm_load_2d(lds0 + nb * (BUF_ELEMS * 2),
                    A + (size_t)m0blk * D_MODEL + k0 + 32,
                    D_MODEL, (unsigned)arows, 32, 128, D_MODEL, 3);
        tdm_load_2d(lds0 + nb * (BUF_ELEMS * 2) + ABUF_ELEMS * 2,
                    B + (size_t)n0 * D_MODEL + k0 + 32,
                    D_MODEL, (unsigned)brows, 32, 64, D_MODEL, 3);
      }
      __builtin_amdgcn_s_wait_tensorcnt(2);   // current tile's 2 ops done
    } else {
      __builtin_amdgcn_s_wait_tensorcnt(0);
    }
    __syncthreads();

    const __bf16* Abuf = lds + buf * BUF_ELEMS;
    const __bf16* Bbuf = Abuf + ABUF_ELEMS;
    // Preload all fragments, then run the WMMA chain (batches ds_loads).
    const v16bf a = load_frag_k32(Abuf + (w * 16 + ln) * LDS_PITCH, 0, hi);
    v16bf bf[4];
#pragma unroll
    for (int nt = 0; nt < 4; ++nt)
      bf[nt] = load_frag_k32(Bbuf + (nt * 16 + ln) * LDS_PITCH, 0, hi);
#pragma unroll
    for (int nt = 0; nt < 4; ++nt) acc[nt] = wmma_bf16(a, bf[nt], acc[nt]);
    __syncthreads();   // all reads of `buf` done before its TDM overwrite
    buf ^= 1;
  }
}

// ---------------------------------------------------------------------------
// 2) Fused QKV projection + RoPE.  grid = (D_MODEL/64, BSROWS/128, 3).
// ---------------------------------------------------------------------------
__global__ __launch_bounds__(256) void qkv_rope_kernel(
    const __bf16* __restrict__ xb,
    const __bf16* __restrict__ wq, const __bf16* __restrict__ wk,
    const __bf16* __restrict__ wv,
    const int* __restrict__ token_pos,
    __bf16* __restrict__ Qb, __bf16* __restrict__ Kb, __bf16* __restrict__ VTb) {
  __shared__ __bf16 lds[2 * BUF_ELEMS];
  const int lane  = threadIdx.x & 31;
  const int w     = threadIdx.x >> 5;
  const int hi    = lane >> 4;
  const int ln    = lane & 15;
  const int which = blockIdx.z;                 // 0=Q 1=K 2=V
  const int n0    = blockIdx.x * 64;
  const int m0    = blockIdx.y * 128 + w * 16;

  const __bf16* W = (which == 0) ? wq : ((which == 1) ? wk : wv);

  v8f acc[4];
#pragma unroll
  for (int nt = 0; nt < 4; ++nt) acc[nt] = vzero8();

  gemm_tn_mainloop(xb, W, blockIdx.y * 128, n0, BSROWS, D_MODEL, lds, lane, w, acc);

  // Epilogue: D-layout row m0 + r + 8*hi, column n0 + nt*16 + ln.
  // RoPE pairs (2i, 2i+1) are adjacent lanes -> shfl_xor(1) gives the partner.
#pragma unroll
  for (int nt = 0; nt < 4; ++nt) {
    const int f     = n0 + nt * 16 + ln;
    const int h_idx = f >> 7;
    const int d_idx = f & 127;
    const float inv_freq = __expf(-((float)(d_idx & ~1) * (1.0f / 128.0f)) * 9.210340371976184f);
    const float sgn = (d_idx & 1) ? 1.0f : -1.0f;
#pragma unroll
    for (int r = 0; r < 8; ++r) {
      const int row   = m0 + r + hi * 8;
      const int b_idx = row >> 11;
      const int s_idx = row & (SEQ - 1);
      const float v = acc[nt][r];
      const float p = __shfl_xor(v, 1, 32);
      float outv = v;
      if (which < 2) {
        const float pos = (float)token_pos[b_idx * SEQ + s_idx];
        const float ang = pos * inv_freq;
        float sn, cs;
        __sincosf(ang, &sn, &cs);
        outv = v * cs + sgn * p * sn;
      }
      if (which == 0) {
        Qb[(((size_t)b_idx * NHEADS + h_idx) * SEQ + s_idx) * HDIM + d_idx] = (__bf16)outv;
      } else if (which == 1) {
        Kb[(((size_t)b_idx * NHEADS + h_idx) * SEQ + s_idx) * HDIM + d_idx] = (__bf16)outv;
      } else {
        VTb[(((size_t)b_idx * NHEADS + h_idx) * HDIM + d_idx) * SEQ + s_idx] = (__bf16)outv;
      }
    }
  }
}

// ---------------------------------------------------------------------------
// 3) Causal flash attention with TDM-staged K/V tiles.
//    grid = (SEQ/128, BATCH*NHEADS), block = 256 (8 waves x 16 q-rows).
//    K tile: 64 rows x 128 d  -> LDS pitch 136 elems (272B: 4r mod 64 banks).
//    V tile: 128 d-rows x 64 s -> LDS pitch 72 elems (144B: 36r mod 64 banks).
// ---------------------------------------------------------------------------
#define KPITCH 136
#define VPITCH 72
#define POFF   0
#define KOFF   (8 * 16 * 64)                  // 8192 elems of P staging first
#define KBUF   (64 * KPITCH)                  // 8704 elems
#define VOFF   (KOFF + 2 * KBUF)              // 25600
#define VBUF   (128 * VPITCH)                 // 9216 elems
#define SMEM_ELEMS (VOFF + 2 * VBUF)          // 44032 elems = 88064 B

__global__ __launch_bounds__(256) void attn_kernel(
    const __bf16* __restrict__ Qb, const __bf16* __restrict__ Kb,
    const __bf16* __restrict__ VTb, __bf16* __restrict__ ctx) {
  __shared__ __bf16 smem[SMEM_ELEMS];

  const int lane  = threadIdx.x & 31;
  const int w     = threadIdx.x >> 5;
  const int hi    = lane >> 4;
  const int ln    = lane & 15;
  const int bh    = blockIdx.y;
  const int b_idx = bh >> 4;
  const int h_idx = bh & 15;
  const int q0    = blockIdx.x * 128;
  const int qr    = q0 + w * 16;

  const __bf16* Qbase = Qb  + (size_t)bh * SEQ * HDIM;
  const __bf16* Kbase = Kb  + (size_t)bh * SEQ * HDIM;
  const __bf16* Vbase = VTb + (size_t)bh * HDIM * SEQ;   // [d][s]
  const unsigned lds0 = (unsigned)(uintptr_t)smem;

  // Q fragments live in registers for the whole kernel (4 K-steps of 32).
  v16bf qf[4];
  const __bf16* qrow = Qbase + (size_t)(qr + ln) * HDIM;
#pragma unroll
  for (int ks = 0; ks < 4; ++ks) qf[ks] = load_frag_k32(qrow, ks * 32, hi);

  v8f acc[8];
#pragma unroll
  for (int dt = 0; dt < 8; ++dt) acc[dt] = vzero8();
  float m8[8], l8[8];
#pragma unroll
  for (int r = 0; r < 8; ++r) { m8[r] = -1e30f; l8[r] = 0.0f; }

  const float sl2e = 1.4426950408889634f * 0.08838834764831845f;  // log2e/sqrt(128)

  const int ntiles = blockIdx.x * 2 + 2;  // causal: k <= q0+127

  if (w == 0) {   // prologue: stage K/V tile 0 into buffer 0
    tdm_load_2d(lds0 + KOFF * 2, Kbase, HDIM, SEQ, 128, 64, HDIM, 5);
    tdm_load_2d(lds0 + VOFF * 2, Vbase, SEQ, HDIM, 64, 128, SEQ, 4);
  }

  int buf = 0;
  for (int jt = 0; jt < ntiles; ++jt) {
    const int j0 = jt * 64;

    if (jt + 1 < ntiles) {
      const int nb = buf ^ 1;
      if (w == 0) {   // prefetch next K/V tile while we do softmax VALU work
        tdm_load_2d(lds0 + (KOFF + nb * KBUF) * 2,
                    Kbase + (size_t)(j0 + 64) * HDIM, HDIM, SEQ, 128, 64, HDIM, 5);
        tdm_load_2d(lds0 + (VOFF + nb * VBUF) * 2,
                    Vbase + (j0 + 64), SEQ, HDIM, 64, 128, SEQ, 4);
      }
      __builtin_amdgcn_s_wait_tensorcnt(2);
    } else {
      __builtin_amdgcn_s_wait_tensorcnt(0);
    }
    __syncthreads();

    const __bf16* kbuf = smem + KOFF + buf * KBUF;
    const __bf16* vbuf = smem + VOFF + buf * VBUF;

    // S = Q * K^T (16x64), 4 N-tiles, K fragments from LDS.
    v8f sacc[4];
#pragma unroll
    for (int nt = 0; nt < 4; ++nt) {
      sacc[nt] = vzero8();
      const __bf16* krow = kbuf + (nt * 16 + ln) * KPITCH;
      v16bf kf[4];
#pragma unroll
      for (int ks = 0; ks < 4; ++ks) kf[ks] = load_frag_k32(krow, ks * 32, hi);
#pragma unroll
      for (int ks = 0; ks < 4; ++ks) sacc[nt] = wmma_bf16(qf[ks], kf[ks], sacc[nt]);
    }

    // Scale + causal mask into exp2 domain.
    float sl[4][8];
#pragma unroll
    for (int nt = 0; nt < 4; ++nt) {
      const int kc = j0 + nt * 16 + ln;
#pragma unroll
      for (int r = 0; r < 8; ++r) {
        const int qg = qr + r + hi * 8;
        const float sv = sacc[nt][r] * sl2e;
        sl[nt][r] = (kc > qg) ? -1e30f : sv;
      }
    }

    // Online softmax update (rows replicated across each 16-lane half).
#pragma unroll
    for (int r = 0; r < 8; ++r) {
      float t = sl[0][r];
#pragma unroll
      for (int nt = 1; nt < 4; ++nt) t = fmaxf(t, sl[nt][r]);
      t = fmaxf(t, __shfl_xor(t, 1, 32));
      t = fmaxf(t, __shfl_xor(t, 2, 32));
      t = fmaxf(t, __shfl_xor(t, 4, 32));
      t = fmaxf(t, __shfl_xor(t, 8, 32));
      const float mnew = fmaxf(m8[r], t);
      const float cor  = exp2f(m8[r] - mnew);
      m8[r] = mnew;
      float rs = 0.0f;
#pragma unroll
      for (int nt = 0; nt < 4; ++nt) {
        const float p = exp2f(sl[nt][r] - mnew);
        sl[nt][r] = p;
        rs += p;
      }
      rs += __shfl_xor(rs, 1, 32);
      rs += __shfl_xor(rs, 2, 32);
      rs += __shfl_xor(rs, 4, 32);
      rs += __shfl_xor(rs, 8, 32);
      l8[r] = l8[r] * cor + rs;
#pragma unroll
      for (int dt = 0; dt < 8; ++dt) acc[dt][r] *= cor;
    }

    // P: D-layout regs -> bf16 LDS -> A-layout fragments (per-wave region).
#pragma unroll
    for (int nt = 0; nt < 4; ++nt)
#pragma unroll
      for (int r = 0; r < 8; ++r)
        smem[POFF + w * 1024 + (r + hi * 8) * 64 + nt * 16 + ln] = (__bf16)sl[nt][r];
    __syncthreads();

    v16bf pf[2];
    const __bf16* prow = smem + POFF + w * 1024 + ln * 64;
#pragma unroll
    for (int ks = 0; ks < 2; ++ks) pf[ks] = load_frag_k32(prow, ks * 32, hi);

    // O += P * V, V fragments from LDS (column j-offset is tile-local).
#pragma unroll
    for (int dt = 0; dt < 8; ++dt) {
      const __bf16* vrow = vbuf + (dt * 16 + ln) * VPITCH;
      const v16bf vf0 = load_frag_k32(vrow, 0, hi);
      const v16bf vf1 = load_frag_k32(vrow, 32, hi);
      acc[dt] = wmma_bf16(pf[0], vf0, acc[dt]);
      acc[dt] = wmma_bf16(pf[1], vf1, acc[dt]);
    }
    __syncthreads();   // done reading this K/V buffer before TDM overwrite
    buf ^= 1;
  }

  // Normalize and write ctx[bs, f] bf16 (row-major, K-major for final GEMM).
#pragma unroll
  for (int dt = 0; dt < 8; ++dt) {
    const int f = h_idx * HDIM + dt * 16 + ln;
#pragma unroll
    for (int r = 0; r < 8; ++r) {
      const int qg = qr + r + hi * 8;
      const float o = acc[dt][r] / l8[r];
      ctx[(size_t)(b_idx * SEQ + qg) * D_MODEL + f] = (__bf16)o;
    }
  }
}

// ---------------------------------------------------------------------------
// 4) Output projection (fp32 out). grid = (D_MODEL/64, BSROWS/128).
// ---------------------------------------------------------------------------
__global__ __launch_bounds__(256) void out_proj_kernel(
    const __bf16* __restrict__ ctx, const __bf16* __restrict__ wo,
    float* __restrict__ out) {
  __shared__ __bf16 lds[2 * BUF_ELEMS];
  const int lane = threadIdx.x & 31;
  const int w    = threadIdx.x >> 5;
  const int hi   = lane >> 4;
  const int ln   = lane & 15;
  const int n0   = blockIdx.x * 64;
  const int m0   = blockIdx.y * 128 + w * 16;

  v8f acc[4];
#pragma unroll
  for (int nt = 0; nt < 4; ++nt) acc[nt] = vzero8();

  gemm_tn_mainloop(ctx, wo, blockIdx.y * 128, n0, BSROWS, D_MODEL, lds, lane, w, acc);

#pragma unroll
  for (int nt = 0; nt < 4; ++nt) {
    const int e = n0 + nt * 16 + ln;
#pragma unroll
    for (int r = 0; r < 8; ++r) {
      const int row = m0 + r + hi * 8;
      out[(size_t)row * D_MODEL + e] = acc[nt][r];
    }
  }
}

// ---------------------------------------------------------------------------
// Host-side orchestration.
// ---------------------------------------------------------------------------
extern "C" void kernel_launch(void* const* d_in, const int* in_sizes, int n_in,
                              void* d_out, int out_size, void* d_ws, size_t ws_size,
                              hipStream_t stream) {
  (void)in_sizes; (void)n_in; (void)out_size; (void)ws_size;

  const float* x  = (const float*)d_in[0];
  const int*   tp = (const int*)d_in[1];
  const float* Wq = (const float*)d_in[2];
  const float* Wk = (const float*)d_in[3];
  const float* Wv = (const float*)d_in[4];
  const float* Wo = (const float*)d_in[5];

  char* ws = (char*)d_ws;
  size_t off = 0;
  __bf16* xb   = (__bf16*)(ws + off); off += (size_t)BSROWS * D_MODEL * 2;
  __bf16* wqb  = (__bf16*)(ws + off); off += (size_t)D_MODEL * D_MODEL * 2;
  __bf16* wkb  = (__bf16*)(ws + off); off += (size_t)D_MODEL * D_MODEL * 2;
  __bf16* wvb  = (__bf16*)(ws + off); off += (size_t)D_MODEL * D_MODEL * 2;
  __bf16* wob  = (__bf16*)(ws + off); off += (size_t)D_MODEL * D_MODEL * 2;
  __bf16* Qb   = (__bf16*)(ws + off); off += (size_t)BSROWS * D_MODEL * 2;
  __bf16* Kb   = (__bf16*)(ws + off); off += (size_t)BSROWS * D_MODEL * 2;
  __bf16* VTb  = (__bf16*)(ws + off); off += (size_t)BSROWS * D_MODEL * 2;
  __bf16* ctxb = (__bf16*)(ws + off); off += (size_t)BSROWS * D_MODEL * 2;

  const int nx = BSROWS * D_MODEL;
  const int nw = D_MODEL * D_MODEL;
  cvt_f32_bf16<<<(nx + 255) / 256, 256, 0, stream>>>(x,  xb,  nx);
  cvt_f32_bf16<<<(nw + 255) / 256, 256, 0, stream>>>(Wq, wqb, nw);
  cvt_f32_bf16<<<(nw + 255) / 256, 256, 0, stream>>>(Wk, wkb, nw);
  cvt_f32_bf16<<<(nw + 255) / 256, 256, 0, stream>>>(Wv, wvb, nw);
  cvt_f32_bf16<<<(nw + 255) / 256, 256, 0, stream>>>(Wo, wob, nw);

  qkv_rope_kernel<<<dim3(D_MODEL / 64, BSROWS / 128, 3), 256, 0, stream>>>(
      xb, wqb, wkb, wvb, tp, Qb, Kb, VTb);

  attn_kernel<<<dim3(SEQ / 128, BATCH * NHEADS), 256, 0, stream>>>(
      Qb, Kb, VTb, ctxb);

  out_proj_kernel<<<dim3(D_MODEL / 64, BSROWS / 128), 256, 0, stream>>>(
      ctxb, wob, (float*)d_out);
}
